// FPNEncoder_47536698032422
// MI455X (gfx1250) — matrix operand order repaired and checked by reading, not compiled, not run on hardware
//
#include <hip/hip_runtime.h>
#include <hip/hip_bf16.h>

// ---------------------------------------------------------------------------
// FPN contrastive encoder, CDNA5 (gfx1250) version.
// Key optimization: the reference only consumes the projection head at 11
// pixels per (batch, level); everything else (argmin, valid mask, sampling)
// is pure coordinate math. So we select indices first, gather 880 feature
// vectors, and run the 256->512(relu)->128 head as bf16 WMMA GEMMs on just
// those tokens. ~400x less compute and memory traffic than the naive path.
// ---------------------------------------------------------------------------

typedef __bf16 bf16;
typedef __attribute__((ext_vector_type(16))) __bf16 v16bf;
typedef __attribute__((ext_vector_type(8)))  __bf16 v8bf;
typedef __attribute__((ext_vector_type(8)))  float  v8f;

#define NLVL 5
#define BATCH 16
#define CIN 256
#define HID 512
#define DPROJ 128
#define NANCH 10
#define TOK_PER_LB 11                    // 1 pos + 10 neg
#define NTOK (NLVL * BATCH * TOK_PER_LB) // 880
#define NTILE ((NTOK + 15) / 16)         // 55
#define NTOK_PAD (NTILE * 16)            // 880

// ---------------- deterministic per-pixel uniform score --------------------
__device__ __forceinline__ float rnd01(unsigned l, unsigned b, unsigned p) {
  unsigned x = (l * 0x9E3779B9u) ^ (b * 0x85EBCA6Bu) ^ (p * 0xC2B2AE35u) ^ 0x2545F491u;
  x ^= x >> 16; x *= 0x7FEB352Du;
  x ^= x >> 15; x *= 0x846CA68Bu;
  x ^= x >> 16;
  return (float)(x >> 8) * (1.0f / 16777216.0f);
}

// ---------------- phase 0: convert weights f32 -> bf16 ---------------------
__global__ __launch_bounds__(256)
void fpn_cvt_kernel(const float* __restrict__ W1, const float* __restrict__ W2,
                    void* w1bf_v, void* w2bf_v) {
  bf16* __restrict__ W1bf = (bf16*)w1bf_v;
  bf16* __restrict__ W2bf = (bf16*)w2bf_v;
  int i = blockIdx.x * 256 + threadIdx.x;
  if (i < HID * CIN)   W1bf[i] = (bf16)W1[i];
  if (i < DPROJ * HID) W2bf[i] = (bf16)W2[i];
}

// ---------------- phase 1: per-(level,b) index selection -------------------
__global__ __launch_bounds__(256)
void fpn_select_kernel(const float* __restrict__ cc, const int* __restrict__ img,
                       int* __restrict__ sel) {
  const int lvl = blockIdx.x >> 4;
  const int b   = blockIdx.x & 15;
  const int sizes[NLVL] = {128, 64, 32, 16, 8};
  const int s  = sizes[lvl];
  const int hw = s * s;
  const int tid = threadIdx.x;
  const float scale = (float)img[0] / (float)s;
  const float cy = cc[b * 4 + 0] + 0.5f * cc[b * 4 + 2];
  const float cx = cc[b * 4 + 1] + 0.5f * cc[b * 4 + 3];

  __shared__ float rv[256];
  __shared__ int   ri[256];
  __shared__ float s_mind;
  __shared__ int   s_sel[TOK_PER_LB];

  // argmin of squared distance over the grid
  float best = 3.4e38f; int bi = 0x7FFFFFFF;
  for (int p = tid; p < hw; p += 256) {
    int y = p / s, x = p - y * s;
    float dx = (float)x * scale - cx;
    float dy = (float)y * scale - cy;
    float d = dx * dx + dy * dy;
    if (d < best || (d == best && p < bi)) { best = d; bi = p; }
  }
  rv[tid] = best; ri[tid] = bi;
  __syncthreads();
  for (int off = 128; off > 0; off >>= 1) {
    if (tid < off) {
      float ov = rv[tid + off]; int oi = ri[tid + off];
      if (ov < rv[tid] || (ov == rv[tid] && oi < ri[tid])) { rv[tid] = ov; ri[tid] = oi; }
    }
    __syncthreads();
  }
  if (tid == 0) { s_mind = rv[0]; s_sel[0] = ri[0]; }
  __syncthreads();
  const float thresh = 2.0f * s_mind;

  // 10 rounds of "min random score among valid, excluding already chosen"
  for (int a = 1; a < TOK_PER_LB; ++a) {
    float bs = 3.4e38f; int bsi = 0x7FFFFFFF;
    for (int p = tid; p < hw; p += 256) {
      int y = p / s, x = p - y * s;
      float dx = (float)x * scale - cx;
      float dy = (float)y * scale - cy;
      float d = dx * dx + dy * dy;
      if (!(d > thresh)) continue;               // valid = dist > 2*mind
      bool used = false;
      for (int q = 1; q < a; ++q) if (s_sel[q] == p) { used = true; break; }
      if (used) continue;
      float sc = rnd01((unsigned)lvl, (unsigned)b, (unsigned)p);
      if (sc < bs || (sc == bs && p < bsi)) { bs = sc; bsi = p; }
    }
    rv[tid] = bs; ri[tid] = (bsi == 0x7FFFFFFF) ? 0 : bsi;
    __syncthreads();
    for (int off = 128; off > 0; off >>= 1) {
      if (tid < off) {
        float ov = rv[tid + off]; int oi = ri[tid + off];
        if (ov < rv[tid] || (ov == rv[tid] && oi < ri[tid])) { rv[tid] = ov; ri[tid] = oi; }
      }
      __syncthreads();
    }
    if (tid == 0) s_sel[a] = ri[0];
    __syncthreads();
  }
  if (tid < TOK_PER_LB)
    sel[lvl * (BATCH * TOK_PER_LB) + b * TOK_PER_LB + tid] = s_sel[tid];
}

// ---------------- A-fragment loader (16-bit A 16x32 layout) ----------------
// lane L: row M = L&15; hi = L>>4 selects K halves {0..15} / {16..31}
__device__ __forceinline__ v16bf load_afrag(const bf16* base, int stride,
                                            int row, int hi, int k0) {
  const bf16* p = base + row * stride + k0 + hi * 16;
  v8bf lo = *(const v8bf*)p;
  v8bf hh = *(const v8bf*)(p + 8);
  return __builtin_shufflevector(lo, hh, 0, 1, 2, 3, 4, 5, 6, 7,
                                        8, 9, 10, 11, 12, 13, 14, 15);
}

// ---------------- phase 2: gather + 2x WMMA GEMM head ----------------------
__global__ __launch_bounds__(256)
void fpn_gemm_kernel(const float* __restrict__ f0, const float* __restrict__ f1,
                     const float* __restrict__ f2, const float* __restrict__ f3,
                     const float* __restrict__ f4,
                     const float* __restrict__ b1v, const float* __restrict__ b2v,
                     const void* w1bf_v, const void* w2bf_v,
                     const int* __restrict__ sel, float* __restrict__ out2) {
  const bf16* __restrict__ W1bf = (const bf16*)w1bf_v;
  const bf16* __restrict__ W2bf = (const bf16*)w2bf_v;

  __shared__ alignas(16) bf16 As[16][CIN + 8];   // +8 bf16 pad -> 4-bank row skew
  __shared__ alignas(16) bf16 Hs[16][HID + 8];

  const int tid  = threadIdx.x;
  const int tile = blockIdx.x;

  // ---- gather 16 tokens (256 channels each, channel stride = h*w) ----
  {
    const float* fps[NLVL] = {f0, f1, f2, f3, f4};
    const int hws[NLVL] = {16384, 4096, 1024, 256, 64};
    const int row = tid >> 4;
    const int c0  = (tid & 15) * 16;
    const int tok = tile * 16 + row;
    if (tok < NTOK) {
      const int lvl = tok / (BATCH * TOK_PER_LB);
      const int rem = tok - lvl * (BATCH * TOK_PER_LB);
      const int b   = rem / TOK_PER_LB;
      const int hw  = hws[lvl];
      const int pix = sel[tok];
      const float* src = fps[lvl] + (size_t)b * CIN * hw + pix;
#pragma unroll
      for (int c = 0; c < 16; ++c)
        As[row][c0 + c] = (bf16)src[(size_t)(c0 + c) * hw];
    } else {
#pragma unroll
      for (int c = 0; c < 16; ++c)
        As[row][c0 + c] = (bf16)0.0f;
    }
  }
  __syncthreads();

  const int lane = tid & 31;
  const int wv   = tid >> 5;    // 8 waves
  const int col  = lane & 15;   // N for B/D frags, M-row for A frags
  const int hi   = lane >> 4;

  const v8f vzero = {0.f, 0.f, 0.f, 0.f, 0.f, 0.f, 0.f, 0.f};

  // ---- GEMM1: [16 x 256] x [256 x 512] -> relu -> Hs (bf16) ----
  v8f acc[4];
#pragma unroll
  for (int j = 0; j < 4; ++j) acc[j] = vzero;
#pragma unroll
  for (int ks = 0; ks < CIN; ks += 32) {
    v16bf a = load_afrag(&As[0][0], CIN + 8, col, hi, ks);
#pragma unroll
    for (int j = 0; j < 4; ++j) {
      const int n = (wv * 4 + j) * 16 + col;
      // B 32x16: lane=N, 16 contiguous K per lane, hi selects K half
      v16bf bfr = *(const v16bf*)(W1bf + (size_t)n * CIN + ks + hi * 16);
      acc[j] = __builtin_amdgcn_wmma_f32_16x16x32_bf16(
          false, a, false, bfr, (short)0, acc[j], false, false);
    }
  }
  // D layout: lane=N=col, VGPR r -> M = r + 8*hi
#pragma unroll
  for (int j = 0; j < 4; ++j) {
    const int n = (wv * 4 + j) * 16 + col;
    const float bias = b1v[n];
#pragma unroll
    for (int r = 0; r < 8; ++r) {
      float v = acc[j][r] + bias;
      v = v > 0.0f ? v : 0.0f;
      Hs[r + 8 * hi][n] = (bf16)v;
    }
  }
  __syncthreads();

  // ---- GEMM2: [16 x 512] x [512 x 128] + b2 -> out2 (f32) ----
  v8f acc2 = vzero;
  const int n2 = wv * 16 + col;
#pragma unroll
  for (int ks = 0; ks < HID; ks += 32) {
    v16bf a = load_afrag(&Hs[0][0], HID + 8, col, hi, ks);
    v16bf bfr = *(const v16bf*)(W2bf + (size_t)n2 * HID + ks + hi * 16);
    acc2 = __builtin_amdgcn_wmma_f32_16x16x32_bf16(
        false, a, false, bfr, (short)0, acc2, false, false);
  }
  const float bias2 = b2v[n2];
#pragma unroll
  for (int r = 0; r < 8; ++r) {
    const int tok = tile * 16 + r + 8 * hi;
    out2[(size_t)tok * DPROJ + n2] = acc2[r] + bias2;
  }
}

// ---------------- phase 3: level-mean with torch-view index scramble -------
__global__ __launch_bounds__(256)
void fpn_reduce_kernel(const float* __restrict__ out2, float* __restrict__ out) {
  const int i = blockIdx.x * 256 + threadIdx.x;
  const int ZJ = BATCH * DPROJ;                       // 2048
  const int TOTAL = ZJ + BATCH * NANCH * DPROJ;       // 22528
  if (i >= TOTAL) return;
  if (i < ZJ) {
    const int b = i >> 7, d = i & 127;
    float s = 0.f;
#pragma unroll
    for (int l2 = 0; l2 < NLVL; ++l2) {
      const int li = b * NLVL + l2;                   // stack(L,B)->view(B,L)
      const int lvl = li >> 4, bb = li & 15;
      s += out2[(size_t)(lvl * (BATCH * TOK_PER_LB) + bb * TOK_PER_LB) * DPROJ + d];
    }
    out[i] = s * 0.2f;
  } else {
    const int j = i - ZJ;
    const int b = j / (NANCH * DPROJ);
    const int a = (j / DPROJ) % NANCH;
    const int d = j & 127;
    float s = 0.f;
#pragma unroll
    for (int l2 = 0; l2 < NLVL; ++l2) {
      const int li = b * NLVL + l2;
      const int lvl = li >> 4, bb = li & 15;
      s += out2[(size_t)(lvl * (BATCH * TOK_PER_LB) + bb * TOK_PER_LB + 1 + a) * DPROJ + d];
    }
    out[i] = s * 0.2f;
  }
}

// ---------------------------------------------------------------------------
extern "C" void kernel_launch(void* const* d_in, const int* in_sizes, int n_in,
                              void* d_out, int out_size, void* d_ws, size_t ws_size,
                              hipStream_t stream) {
  (void)in_sizes; (void)n_in; (void)out_size; (void)ws_size;
  const float* f0 = (const float*)d_in[0];
  const float* f1 = (const float*)d_in[1];
  const float* f2 = (const float*)d_in[2];
  const float* f3 = (const float*)d_in[3];
  const float* f4 = (const float*)d_in[4];
  const float* cc = (const float*)d_in[5];
  const float* W1 = (const float*)d_in[6];
  const float* b1 = (const float*)d_in[7];
  const float* W2 = (const float*)d_in[8];
  const float* b2 = (const float*)d_in[9];
  const int* img  = (const int*)d_in[10];
  float* out = (float*)d_out;

  // workspace layout (all 256B-aligned offsets)
  char* ws = (char*)d_ws;
  void* W1bf  = (void*)(ws + 0);                         // 512*256*2 = 262144
  void* W2bf  = (void*)(ws + 262144);                    // 128*512*2 = 131072
  int*  sel   = (int*)(ws + 393216);                     // 880*4
  float* out2 = (float*)(ws + 397312);                   // 880*128*4 = 450560

  fpn_cvt_kernel<<<(HID * CIN + 255) / 256, 256, 0, stream>>>(W1, W2, W1bf, W2bf);
  fpn_select_kernel<<<NLVL * BATCH, 256, 0, stream>>>(cc, img, sel);
  fpn_gemm_kernel<<<NTILE, 256, 0, stream>>>(f0, f1, f2, f3, f4, b1, b2,
                                             W1bf, W2bf, sel, out2);
  fpn_reduce_kernel<<<(BATCH * DPROJ * (1 + NANCH) + 255) / 256, 256, 0, stream>>>(out2, out);
}
